// MultiHeadAttention_266287972474
// MI455X (gfx1250) — matrix operand order repaired and checked by reading, not compiled, not run on hardware
//
#include <hip/hip_runtime.h>

#define Bb 16
#define Ee 256
#define Nn 512
#define Hh 8
#define Dd 32
#define SROW 516   // padded f32 score row stride (516 % 64 == 4 -> staggered banks)
#define APAD 34    // bf16 row stride for [*][32] K-major tiles (68B = 17 dwords, odd -> staggered)
#define VPAD 514   // bf16 row stride for [32][512] m-major V tiles (1028B = 257 dwords, odd)

typedef __attribute__((ext_vector_type(16))) __bf16 v16bf;
typedef __attribute__((ext_vector_type(8)))  float  v8f;

static __device__ __forceinline__ v8f wmma_bf16(v16bf a, v16bf b, v8f c) {
  // D(16x16 f32) = A(16x32 bf16) * B(32x16 bf16) + C
  return __builtin_amdgcn_wmma_f32_16x16x32_bf16(false, a, false, b, (short)0, c,
                                                 false, false);
}

static __device__ __forceinline__ void bsplit(float x, __bf16& h, __bf16& l) {
  h = (__bf16)x;
  l = (__bf16)(x - (float)h);
}

// A-frag (16x32 bf16, row-major LDS row of >=32 bf16). Element i, half-wave hi8=hi*8:
//   j=i/2, p=i%2 : K = hi8+2j+p (j<4) | 16+hi8+2(j-4)+p (j>=4)
static __device__ __forceinline__ v16bf load_afrag(const __bf16* row, int hi8) {
  v16bf a;
#pragma unroll
  for (int i = 0; i < 16; ++i) {
    int j = i >> 1, p = i & 1;
    int k = (j < 4) ? (hi8 + 2 * j + p) : (16 + hi8 + 2 * (j - 4) + p);
    a[i] = row[k];
  }
  return a;
}

// B-frag (32x16 bf16): element i is K = hi8 + i (i<8) | 16 + hi8 + (i-8) (i>=8), hi8 = hi16/2, at fixed N.
static __device__ __forceinline__ v16bf load_bfrag(const __bf16* base, int hi16) {
  v16bf b;
#pragma unroll
  for (int i = 0; i < 16; ++i) b[i] = base[(i < 8) ? ((hi16 >> 1) + i) : (16 + (hi16 >> 1) + (i - 8))];
  return b;
}

// Y[b,o,n] = sum_e W[o,e] * X[b,e,n] + bias[o]   (split-bf16 x3 WMMA, f32 accumulate)
// grid: (N/64, E/64, B), block 128 (4 waves); wave w owns o-rows [oTile+16w,+16) x 64 n.
__global__ __launch_bounds__(128) void proj_kernel(const float* __restrict__ X,
                                                   const float* __restrict__ W,
                                                   const float* __restrict__ bias,
                                                   float* __restrict__ Y) {
  __shared__ __bf16 Wh[64][APAD], Wl[64][APAD];  // A tile: [o][k], k-slab of 32
  __shared__ __bf16 Xh[64][APAD], Xl[64][APAD];  // B tile transposed: [n][k]
  const int nTile = blockIdx.x * 64;
  const int oTile = blockIdx.y * 64;
  const int b     = blockIdx.z;
  const int tid   = threadIdx.x;
  const int wave  = tid >> 5;
  const int lane  = tid & 31;
  const int l16   = lane & 15;
  const int hi    = lane >> 4;
  const int hi8   = hi * 8, hi16 = hi * 16;

  v8f acc[4];
#pragma unroll
  for (int j = 0; j < 4; ++j) acc[j] = (v8f){0, 0, 0, 0, 0, 0, 0, 0};

  for (int e0 = 0; e0 < Ee; e0 += 32) {
    // stage W 64x32 (512 float4, 4 per thread), split hi/lo
#pragma unroll
    for (int it = 0; it < 4; ++it) {
      int slot = tid + it * 128;      // 0..511
      int row  = slot >> 3;           // 0..63
      int col  = (slot & 7) * 4;
      float4 w4 = *(const float4*)(W + (size_t)(oTile + row) * Ee + e0 + col);
      bsplit(w4.x, Wh[row][col + 0], Wl[row][col + 0]);
      bsplit(w4.y, Wh[row][col + 1], Wl[row][col + 1]);
      bsplit(w4.z, Wh[row][col + 2], Wl[row][col + 2]);
      bsplit(w4.w, Wh[row][col + 3], Wl[row][col + 3]);
    }
    // stage X 32x64 -> transposed [n][k], coalesced along n
#pragma unroll
    for (int it = 0; it < 4; ++it) {
      int slot = tid + it * 128;
      int krow = slot >> 4;           // 0..31
      int coln = (slot & 15) * 4;
      float4 x4 = *(const float4*)(X + ((size_t)b * Ee + e0 + krow) * Nn + nTile + coln);
      bsplit(x4.x, Xh[coln + 0][krow], Xl[coln + 0][krow]);
      bsplit(x4.y, Xh[coln + 1][krow], Xl[coln + 1][krow]);
      bsplit(x4.z, Xh[coln + 2][krow], Xl[coln + 2][krow]);
      bsplit(x4.w, Xh[coln + 3][krow], Xl[coln + 3][krow]);
    }
    __syncthreads();

    const int mrow = wave * 16 + l16;
    v16bf ah = load_afrag(&Wh[mrow][0], hi8);
    v16bf al = load_afrag(&Wl[mrow][0], hi8);
#pragma unroll
    for (int j = 0; j < 4; ++j) {
      v16bf bh = load_bfrag(&Xh[j * 16 + l16][0], hi16);
      v16bf bl = load_bfrag(&Xl[j * 16 + l16][0], hi16);
      acc[j] = wmma_bf16(ah, bh, acc[j]);   // hi*hi
      acc[j] = wmma_bf16(ah, bl, acc[j]);   // hi*lo
      acc[j] = wmma_bf16(al, bh, acc[j]);   // lo*hi
    }
    __syncthreads();
  }

  // epilogue: C layout -> (lane, vgpr r) = M = r + 8*hi, N = l16
#pragma unroll
  for (int r = 0; r < 8; ++r) {
    int o = oTile + wave * 16 + r + hi * 8;
    float bv = bias[o];
#pragma unroll
    for (int j = 0; j < 4; ++j) {
      Y[((size_t)b * Ee + o) * Nn + nTile + j * 16 + l16] = acc[j][r] + bv;
    }
  }
}

// One block per (16 query rows, head h, batch b). 128 threads = 4 waves.
__global__ __launch_bounds__(128) void attn_kernel(const float* __restrict__ qh,
                                                   const float* __restrict__ kh,
                                                   const float* __restrict__ vh,
                                                   const float* __restrict__ adj,
                                                   float* __restrict__ oh) {
  extern __shared__ char smemraw[];
  __bf16* Qh  = (__bf16*)smemraw;        // [16][APAD]  q tile hi, [n][d]
  __bf16* Ql  = Qh + 16 * APAD;          // q tile lo
  __bf16* Kth = Ql + 16 * APAD;          // [512][APAD] K panel hi, [m][d]
  __bf16* Ktl = Kth + 512 * APAD;        // K panel lo
  __bf16* Vth = Ktl + 512 * APAD;        // [32][VPAD]  V panel hi, [d][m]
  __bf16* Vtl = Vth + 32 * VPAD;         // V panel lo
  float*  s_sh = (float*)(Vtl + 32 * VPAD);  // [16][SROW] scores -> attn (f32)
  float*  red  = s_sh + 16 * SROW;           // [16][8]
  float*  rowstat = red + 16 * 8;            // [16]

  const int nbase = blockIdx.x * 16;
  const int h     = blockIdx.y;
  const int b     = blockIdx.z;
  const int tid   = threadIdx.x;
  const int wave  = tid >> 5;
  const int lane  = tid & 31;
  const int l16   = lane & 15;
  const int hi    = lane >> 4;
  const int hi8   = hi * 8, hi16 = hi * 16;
  const float NEGINF = -__builtin_inff();

  const size_t headOff = ((size_t)b * Ee + h * Dd) * Nn;

  // ---- phase 1: stage Q (16x32), K panel (512x32), V panel (512x32) as split bf16
  {
    int d  = tid >> 2;           // 0..31
    int r4 = (tid & 3) * 4;      // 0,4,8,12
    const float* qp = qh + headOff + (size_t)d * Nn + nbase + r4;
#pragma unroll
    for (int i = 0; i < 4; ++i)
      bsplit(qp[i], Qh[(r4 + i) * APAD + d], Ql[(r4 + i) * APAD + d]);

    int mb = (tid & 3) * 128;
    const float* kp = kh + headOff + (size_t)d * Nn + mb;
    const float* vp = vh + headOff + (size_t)d * Nn + mb;
    for (int i = 0; i < 128; i += 4) {
      float4 k4 = *(const float4*)(kp + i);
      float4 v4 = *(const float4*)(vp + i);
      int m = mb + i;
      bsplit(k4.x, Kth[(m + 0) * APAD + d], Ktl[(m + 0) * APAD + d]);
      bsplit(k4.y, Kth[(m + 1) * APAD + d], Ktl[(m + 1) * APAD + d]);
      bsplit(k4.z, Kth[(m + 2) * APAD + d], Ktl[(m + 2) * APAD + d]);
      bsplit(k4.w, Kth[(m + 3) * APAD + d], Ktl[(m + 3) * APAD + d]);
      bsplit(v4.x, Vth[d * VPAD + m + 0], Vtl[d * VPAD + m + 0]);
      bsplit(v4.y, Vth[d * VPAD + m + 1], Vtl[d * VPAD + m + 1]);
      bsplit(v4.z, Vth[d * VPAD + m + 2], Vtl[d * VPAD + m + 2]);
      bsplit(v4.w, Vth[d * VPAD + m + 3], Vtl[d * VPAD + m + 3]);
    }
  }
  __syncthreads();

  // ---- phase 2: S = Q K^T, one K=32 step -> 3 split-bf16 WMMAs per 16x16 tile
  {
    v16bf qah = load_afrag(Qh + l16 * APAD, hi8);
    v16bf qal = load_afrag(Ql + l16 * APAD, hi8);
    for (int mt = wave; mt < 32; mt += 4) {
      const int mbase = mt * 16;
      v16bf kbh = load_bfrag(Kth + (size_t)(mbase + l16) * APAD, hi16);
      v16bf kbl = load_bfrag(Ktl + (size_t)(mbase + l16) * APAD, hi16);
      v8f acc = (v8f){0, 0, 0, 0, 0, 0, 0, 0};
      acc = wmma_bf16(qah, kbh, acc);
      acc = wmma_bf16(qah, kbl, acc);
      acc = wmma_bf16(qal, kbh, acc);
#pragma unroll
      for (int r = 0; r < 8; ++r) {
        int nrow = r + hi * 8;
        int n = nbase + nrow;
        int m = mbase + l16;
        float adjv = adj[((size_t)b * Nn + n) * Nn + m];
        float sv = acc[r] * adjv;
        s_sh[nrow * SROW + m] = (adjv != 0.0f) ? sv : NEGINF;
      }
    }
  }
  __syncthreads();

  // ---- phase 3: masked softmax per row (16 rows x 8 threads each), f32
  {
    const int row = tid >> 3;
    const int sub = tid & 7;
    float* srow = s_sh + row * SROW;

    float pmax = NEGINF;
    for (int c = sub; c < Nn; c += 8) pmax = fmaxf(pmax, srow[c]);
    red[row * 8 + sub] = pmax;
    __syncthreads();
    if (sub == 0) {
      float m = red[row * 8];
#pragma unroll
      for (int i = 1; i < 8; ++i) m = fmaxf(m, red[row * 8 + i]);
      rowstat[row] = (m == NEGINF) ? 0.0f : m;   // m_safe
    }
    __syncthreads();
    const float msafe = rowstat[row];
    float psum = 0.0f;
    for (int c = sub; c < Nn; c += 8) {
      float sv = srow[c];
      float e = (sv == NEGINF) ? 0.0f : __expf(sv - msafe);
      srow[c] = e;
      psum += e;
    }
    red[row * 8 + sub] = psum;
    __syncthreads();
    if (sub == 0) {
      float s = 0.0f;
#pragma unroll
      for (int i = 0; i < 8; ++i) s += red[row * 8 + i];
      rowstat[row] = (s > 0.0f) ? (1.0f / s) : 0.0f;  // fully-masked row -> 0
    }
    __syncthreads();
    const float rs = rowstat[row];
    for (int c = sub; c < Nn; c += 8) srow[c] *= rs;
  }
  __syncthreads();

  // ---- phase 4: O = attn(16x512) x V(512x32); waves 0/1 own a 16-wide d tile
  if (wave < 2) {
    const int dbase = wave * 16;
    v8f acc = (v8f){0, 0, 0, 0, 0, 0, 0, 0};
    for (int kt = 0; kt < Nn / 32; ++kt) {
      const int k0 = kt * 32;
      v16bf ah, al;
#pragma unroll
      for (int i = 0; i < 16; ++i) {
        int j = i >> 1, p = i & 1;
        int k = k0 + ((j < 4) ? (hi8 + 2 * j + p) : (16 + hi8 + 2 * (j - 4) + p));
        float x = s_sh[l16 * SROW + k];
        __bf16 xh = (__bf16)x;
        ah[i] = xh;
        al[i] = (__bf16)(x - (float)xh);
      }
      v16bf bh = load_bfrag(Vth + (size_t)(dbase + l16) * VPAD + k0, hi16);
      v16bf bl = load_bfrag(Vtl + (size_t)(dbase + l16) * VPAD + k0, hi16);
      acc = wmma_bf16(ah, bh, acc);
      acc = wmma_bf16(ah, bl, acc);
      acc = wmma_bf16(al, bh, acc);
    }
#pragma unroll
    for (int r = 0; r < 8; ++r) {
      int nrow = r + hi * 8;
      int d = dbase + l16;
      oh[headOff + (size_t)d * Nn + nbase + nrow] = acc[r];
    }
  }
}

extern "C" void kernel_launch(void* const* d_in, const int* in_sizes, int n_in,
                              void* d_out, int out_size, void* d_ws, size_t ws_size,
                              hipStream_t stream) {
  const float* q   = (const float*)d_in[0];
  const float* k   = (const float*)d_in[1];
  const float* v   = (const float*)d_in[2];
  const float* adj = (const float*)d_in[3];
  const float* Wq  = (const float*)d_in[4];
  const float* bq  = (const float*)d_in[5];
  const float* Wk  = (const float*)d_in[6];
  const float* bk  = (const float*)d_in[7];
  const float* Wv  = (const float*)d_in[8];
  const float* bv  = (const float*)d_in[9];
  const float* Wo  = (const float*)d_in[10];
  const float* bo  = (const float*)d_in[11];
  float* out = (float*)d_out;

  const size_t planeSz = (size_t)Bb * Ee * Nn;
  float* qh = (float*)d_ws;
  float* kh = qh + planeSz;
  float* vh = kh + planeSz;
  float* oh = vh + planeSz;

  // attn dynamic LDS: bf16 panels + f32 score buffer
  const size_t smem = (size_t)(2 * 16 * APAD + 2 * 512 * APAD + 2 * 32 * VPAD) * 2 +
                      (size_t)(16 * SROW + 16 * 8 + 16) * 4;
  hipFuncSetAttribute((const void*)attn_kernel,
                      hipFuncAttributeMaxDynamicSharedMemorySize, (int)smem);

  dim3 pgrid(Nn / 64, Ee / 64, Bb);
  proj_kernel<<<pgrid, 128, 0, stream>>>(q, Wq, bq, qh);
  proj_kernel<<<pgrid, 128, 0, stream>>>(k, Wk, bk, kh);
  proj_kernel<<<pgrid, 128, 0, stream>>>(v, Wv, bv, vh);

  dim3 agrid(Nn / 16, Hh, Bb);
  attn_kernel<<<agrid, 128, smem, stream>>>(qh, kh, vh, adj, oh);

  proj_kernel<<<pgrid, 128, 0, stream>>>(oh, Wo, bo, out);
}
